// Pointnet2_19439021982076
// MI455X (gfx1250) — matrix operand order repaired and checked by reading, not compiled, not run on hardware
//
#include <hip/hip_runtime.h>
#include <hip/hip_bf16.h>

// ---------------------------------------------------------------------------
// PointNet++ double set-abstraction for MI455X (gfx1250, wave32, WMMA).
//
// FPS fully LDS-resident per batch (serial-latency bound); ball query
// thread-per-query out of L2; both MLP GEMMs via v_wmma_f32_16x16x32_f16
// with f32 accumulation; BN via stats pass + finalize pass (bias folded into
// BN shift); max-over-k fused into the finalize pass. Layer-2 A tiles are
// staged via GLOBAL_LOAD_ASYNC_TO_LDS_B128 (ASYNCcnt) when available.
// ---------------------------------------------------------------------------

typedef __attribute__((ext_vector_type(16))) _Float16 v16h;
typedef __attribute__((ext_vector_type(8)))  float    v8f;

union HV { v16h v; unsigned u[8]; _Float16 h[16]; };

#if defined(__has_builtin)
#if __has_builtin(__builtin_amdgcn_global_load_async_to_lds_b128)
#define HAVE_ASYNC_LDS 1
#endif
#endif
#ifndef HAVE_ASYNC_LDS
#define HAVE_ASYNC_LDS 0
#endif

#if HAVE_ASYNC_LDS
// Exact param types per hipcc diagnostic: vector_size(16) int, AS1 src / AS3 dst.
typedef int v4i __attribute__((vector_size(4 * sizeof(int))));
typedef __attribute__((address_space(1))) v4i* as1_v4i;
typedef __attribute__((address_space(3))) v4i* as3_v4i;
#endif

// Problem constants (match reference).
#define B_    4
#define N1    8192
#define NP1   2048
#define NS1   32
#define CH1   128      // layer-1 out channels (= padded)
#define KP1   32       // layer-1 K padded (6 -> 32)
#define NP2   256
#define NS2   16
#define C2    131      // layer-2 in channels
#define D2    693      // layer-2 out channels
#define KP2   160      // layer-2 K padded (131 -> 160, 5 WMMA K-steps)
#define NPAD2 704      // layer-2 N padded (693 -> 704, 44 tiles)

// ---- WMMA fragment loaders -------------------------------------------------
// A: 16x32 f16 tile from f16 LDS, row-major with row stride KPAD halfs.
// ISA layout: lanes 0-15 carry M=lane&15, VGPR v<4 -> K = 8*hi + 2v (+pair),
//             VGPR v>=4 -> K = 16 + 8*hi + 2(v-4); hi = lane>=16.
static __device__ inline v16h load_a_frag(const _Float16* lds, int rowBase,
                                          int KPAD, int kstep, int lane) {
  const int m  = lane & 15;
  const int hi = (lane >> 4) & 1;
  const _Float16* row = lds + (rowBase + m) * KPAD + kstep * 32;
  HV r;
#pragma unroll
  for (int v = 0; v < 4; ++v) {
    r.u[v]     = *(const unsigned*)(row + 8 * hi + 2 * v);
    r.u[v + 4] = *(const unsigned*)(row + 16 + 8 * hi + 2 * v);
  }
  return r.v;
}

// Same A fragment but from an f32 LDS tile (async-staged); convert at load.
static __device__ inline v16h load_a_frag_f32(const float* lds, int rowBase,
                                              int KPAD, int kstep, int lane) {
  const int m  = lane & 15;
  const int hi = (lane >> 4) & 1;
  const float* row = lds + (rowBase + m) * KPAD + kstep * 32;
  HV r;
#pragma unroll
  for (int v = 0; v < 4; ++v) {
    const int k0 = 8 * hi + 2 * v;
    const int k1 = 16 + 8 * hi + 2 * v;
    r.h[2 * v]     = (_Float16)row[k0];
    r.h[2 * v + 1] = (_Float16)row[k0 + 1];
    r.h[8 + 2 * v]     = (_Float16)row[k1];
    r.h[8 + 2 * v + 1] = (_Float16)row[k1 + 1];
  }
  return r.v;
}

// B: 32x16 f16 tile from global weights stored N-major (col n, stride KPAD).
// lane carries column n=lane&15; same K order as A: VGPR v<4 -> K = 8*hi + 2v
// (+pair), VGPR v>=4 -> K = 16 + 8*hi + 2(v-4); hi = lane>=16.
static __device__ inline v16h load_b_frag(const _Float16* w, int ncol0,
                                          int KPAD, int kstep, int lane) {
  const int n    = ncol0 + (lane & 15);
  const int hi   = (lane >> 4) & 1;
  const _Float16* col = w + (size_t)n * KPAD + kstep * 32;
  HV r;
#pragma unroll
  for (int v = 0; v < 4; ++v) {
    r.u[v]     = *(const unsigned*)(col + 8 * hi + 2 * v);
    r.u[v + 4] = *(const unsigned*)(col + 16 + 8 * hi + 2 * v);
  }
  return r.v;
}

// ---- Farthest point sampling: one workgroup per batch, all state in LDS ----
__global__ void fps_kernel(const float* __restrict__ pts, int stride, int n,
                           int npoint, int* __restrict__ out) {
  extern __shared__ float sm[];
  float* X = sm;           float* Y = sm + n;
  float* Z = sm + 2 * n;   float* D = sm + 3 * n;
  float* rv = sm + 4 * n;
  int*   ri = (int*)(rv + blockDim.x);
  __shared__ int s_far;
  const int b = blockIdx.x, tid = threadIdx.x, bd = blockDim.x;
  const float* p = pts + (size_t)b * n * stride;
  for (int i = tid; i < n; i += bd) {
    X[i] = p[i * stride + 0];
    Y[i] = p[i * stride + 1];
    Z[i] = p[i * stride + 2];
    D[i] = 1e10f;
  }
  if (tid == 0) s_far = 0;
  __syncthreads();
  for (int it = 0; it < npoint; ++it) {
    const int far = s_far;
    if (tid == 0) out[b * npoint + it] = far;
    const float cx = X[far], cy = Y[far], cz = Z[far];
    float bm = -1.0f; int bi = 0;
    for (int i = tid; i < n; i += bd) {
      float dx = X[i] - cx, dy = Y[i] - cy, dz = Z[i] - cz;
      float d  = dx * dx + dy * dy + dz * dz;
      float nd = fminf(D[i], d);
      D[i] = nd;
      if (nd > bm) { bm = nd; bi = i; }   // strict > keeps smallest i per thread
    }
    rv[tid] = bm; ri[tid] = bi;
    __syncthreads();
    for (int off = bd >> 1; off > 0; off >>= 1) {
      if (tid < off) {
        float ov = rv[tid + off]; int oi = ri[tid + off];
        if (ov > rv[tid] || (ov == rv[tid] && oi < ri[tid])) {
          rv[tid] = ov; ri[tid] = oi;
        }
      }
      __syncthreads();
    }
    if (tid == 0) s_far = ri[0];
    __syncthreads();
  }
}

// ---- gather xyz of sampled centroids --------------------------------------
__global__ void gather3_kernel(const float* __restrict__ pts, int stride,
                               int nsrc, const int* __restrict__ idx,
                               int count, int total, float* __restrict__ out) {
  int t = blockIdx.x * blockDim.x + threadIdx.x;
  if (t >= total) return;
  int b = t / count;
  int j = idx[t];
  const float* p = pts + ((size_t)(b * nsrc + j)) * stride;
  out[t * 3 + 0] = p[0];
  out[t * 3 + 1] = p[1];
  out[t * 3 + 2] = p[2];
}

// ---- ball query: first nsample in-index-order within radius, pad w/ first --
__global__ void ball_query_kernel(const float* __restrict__ pts, int stride,
                                  int nsrc, const float* __restrict__ nq,
                                  int count, int total, int nsample, float r2,
                                  int* __restrict__ out) {
  int t = blockIdx.x * blockDim.x + threadIdx.x;
  if (t >= total) return;
  int b = t / count;
  const float* p = pts + (size_t)b * nsrc * stride;
  float qx = nq[t * 3 + 0], qy = nq[t * 3 + 1], qz = nq[t * 3 + 2];
  int* o = out + (size_t)t * nsample;
  int cnt = 0, first = 0; bool have = false;
  for (int i = 0; i < nsrc; ++i) {
    float dx = p[i * stride + 0] - qx;
    float dy = p[i * stride + 1] - qy;
    float dz = p[i * stride + 2] - qz;
    if (dx * dx + dy * dy + dz * dz <= r2) {
      if (!have) { first = i; have = true; }
      o[cnt++] = i;
      if (cnt == nsample) break;
    }
  }
  for (int c = cnt; c < nsample; ++c) o[c] = first;
}

// ---- pack weights f32 (C x D) -> f16 N-major padded (NPAD x KPAD) ----------
// l2order: layer-2 K permutation [feat(0..127)=orig c 3..130, xyz=orig c 0..2]
// so async-staged feature rows land at 16B-aligned K=0 in LDS.
__global__ void pack_w_kernel(const float* __restrict__ w, int C, int D,
                              int KPAD, int NPAD, int l2order,
                              _Float16* __restrict__ out) {
  int t = blockIdx.x * blockDim.x + threadIdx.x;
  if (t >= NPAD * KPAD) return;
  int n = t / KPAD, k = t % KPAD;
  int c;
  if (l2order) c = (k < 128) ? (k + 3) : ((k < 131) ? (k - 128) : -1);
  else         c = (k < C) ? k : -1;
  float v = (c >= 0 && n < D) ? w[c * D + n] : 0.0f;
  out[t] = (_Float16)v;
}

__global__ void zero_f32(float* __restrict__ p, int n) {
  int t = blockIdx.x * blockDim.x + threadIdx.x;
  if (t < n) p[t] = 0.0f;
}

// ---- BN scale/shift from accumulated stats; GEMM bias folded into shift ----
__global__ void bn_finalize(const float* __restrict__ stats,
                            const float* __restrict__ gamma,
                            const float* __restrict__ beta,
                            const float* __restrict__ bias, float invcnt,
                            int nch, int npad, float* __restrict__ scale,
                            float* __restrict__ shift) {
  int t = blockIdx.x * blockDim.x + threadIdx.x;
  if (t >= npad) return;
  if (t < nch) {
    float mu  = stats[t] * invcnt;
    float var = stats[npad + t] * invcnt - mu * mu;
    float inv = rsqrtf(var + 1e-5f);
    float sc  = gamma[t] * inv;
    scale[t]  = sc;
    shift[t]  = beta[t] - mu * sc + bias[t] * sc;  // h = gemm + bias
  } else {
    scale[t] = 0.0f; shift[t] = 0.0f;
  }
}

// ---- Layer 1 GEMM (262144x6 * 6x128) + BN stats / BN+ReLU+max over k=32 ----
// 128 rows per block (4 s-groups), 8 waves, 8 N-tiles, 1 K-step.
template <bool STATS>
__global__ void sa1_kernel(const float* __restrict__ xyz,
                           const float* __restrict__ new_xyz,
                           const int* __restrict__ idx,
                           const _Float16* __restrict__ wp,
                           const float* __restrict__ bias,
                           float* __restrict__ gstats,
                           const float* __restrict__ scale,
                           const float* __restrict__ shift,
                           float* __restrict__ outp) {
  extern __shared__ char smraw[];
  _Float16* A  = (_Float16*)smraw;                       // 128 x 32 halfs
  float* faux  = (float*)(smraw + 128 * KP1 * sizeof(_Float16));
  // STATS: faux[0..127]=sum, faux[128..255]=sumsq. FINAL: faux = h[128][128].
  const int tid = threadIdx.x;
  const int rowBase = blockIdx.x * 128;
  if (STATS && tid < 128) { faux[tid] = 0.0f; faux[128 + tid] = 0.0f; }
  if (tid < 128) {
    int rg    = rowBase + tid;        // flat (b,s,k) row == idx1 index
    int j     = idx[rg];
    int b     = rg >> 16;             // / (2048*32)
    int sflat = rg >> 5;              // b*2048 + s
    const float* pc = xyz + ((size_t)(b * N1 + j)) * 6;
    const float* ct = new_xyz + (size_t)sflat * 3;
    _Float16* ar = A + tid * KP1;
    ar[0] = (_Float16)(pc[0] - ct[0]);
    ar[1] = (_Float16)(pc[1] - ct[1]);
    ar[2] = (_Float16)(pc[2] - ct[2]);
    ar[3] = (_Float16)pc[3];
    ar[4] = (_Float16)pc[4];
    ar[5] = (_Float16)pc[5];
#pragma unroll
    for (int c = 6; c < KP1; ++c) ar[c] = (_Float16)0.0f;
  }
  __syncthreads();
  const int wave = tid >> 5, lane = tid & 31;
  const v16h af = load_a_frag(A, wave * 16, KP1, 0, lane);
#pragma unroll
  for (int nt = 0; nt < 8; ++nt) {
    v16h bf = load_b_frag(wp, nt * 16, KP1, 0, lane);
    v8f acc = {};
    acc = __builtin_amdgcn_wmma_f32_16x16x32_f16(false, af, false, bf,
                                                 (short)0, acc, false, false);
    const int ch = nt * 16 + (lane & 15);
    if (STATS) {
      float bv = bias[ch];
      float s = 0.0f, q = 0.0f;
#pragma unroll
      for (int i = 0; i < 8; ++i) { float h = acc[i] + bv; s += h; q += h * h; }
      atomicAdd(&faux[ch], s);
      atomicAdd(&faux[128 + ch], q);
    } else {
      float sc = scale[ch], sh = shift[ch];
      const int hi8 = (lane & 16) ? 8 : 0;   // C/D layout: lanes>=16 -> M+8
#pragma unroll
      for (int i = 0; i < 8; ++i) {
        float h = fmaxf(acc[i] * sc + sh, 0.0f);
        faux[(wave * 16 + hi8 + i) * CH1 + ch] = h;
      }
    }
  }
  __syncthreads();
  if (STATS) {
    if (tid < 128) {
      atomicAdd(&gstats[tid], faux[tid]);
      atomicAdd(&gstats[128 + tid], faux[128 + tid]);
    }
  } else {
    // max over k=32 : 4 groups x 128 channels per block
#pragma unroll
    for (int o = tid; o < 512; o += 256) {
      int g = o >> 7, ch = o & 127;
      float m = -1e30f;
      for (int r = 0; r < NS1; ++r) m = fmaxf(m, faux[(g * NS1 + r) * CH1 + ch]);
      int sflat = blockIdx.x * 4 + g;
      outp[(size_t)sflat * CH1 + ch] = m;
    }
  }
}

// ---- Layer 2 GEMM (16384x131 * 131x693) + BN stats / BN+ReLU+max over k=16 -
// 128 rows per block (8 s-groups == 8 waves), 44 N-tiles, 5 K-steps.
// A tile is f32 in LDS: the 128-feature block per row is staged with
// GLOBAL_LOAD_ASYNC_TO_LDS_B128 (ASYNCcnt) when the builtin exists; rel-xyz
// and padding are staged by VALU. K-order: [feats 0..127 | dx dy dz | pad].
template <bool STATS>
__global__ void sa2_kernel(const float* __restrict__ l1xyz,
                           const float* __restrict__ l1pts,
                           const float* __restrict__ new_xyz,
                           const int* __restrict__ idx,
                           const _Float16* __restrict__ wp,
                           const float* __restrict__ bias,
                           float* __restrict__ gstats,
                           const float* __restrict__ scale,
                           const float* __restrict__ shift,
                           float* __restrict__ outp) {
  extern __shared__ char smraw[];
  float* Af   = (float*)smraw;                           // 128 x 160 f32
  float* ssum = (float*)(smraw + 128 * KP2 * sizeof(float)); // 2x704 (STATS)
  const int tid = threadIdx.x;
  const int rowBase = blockIdx.x * 128;
  if (STATS) for (int c = tid; c < 2 * NPAD2; c += 256) ssum[c] = 0.0f;
  // rel-xyz + zero padding (K = 128..159)
  if (tid < 128) {
    int rg    = rowBase + tid;
    int j     = idx[rg];
    int b     = rg >> 12;             // / (256*16)
    int sflat = rg >> 4;              // b*256 + s
    const float* pc = l1xyz + ((size_t)(b * NP1 + j)) * 3;
    const float* ct = new_xyz + (size_t)sflat * 3;
    float* ar = Af + tid * KP2;
    ar[128] = pc[0] - ct[0];
    ar[129] = pc[1] - ct[1];
    ar[130] = pc[2] - ct[2];
#pragma unroll
    for (int c = C2; c < KP2; ++c) ar[c] = 0.0f;
  }
  // feature block (K = 0..127): 128 rows x 32 x 16B chunks
#if HAVE_ASYNC_LDS
  for (int c = tid; c < 128 * 32; c += 256) {
    int row = c >> 5, q = c & 31;
    int rg  = rowBase + row;
    int j   = idx[rg];
    int b   = rg >> 12;
    const float* src = l1pts + ((size_t)(b * NP1 + j)) * CH1 + q * 4;
    float* dst = Af + row * KP2 + q * 4;
    __builtin_amdgcn_global_load_async_to_lds_b128(
        (as1_v4i)(size_t)src, (as3_v4i)(size_t)dst, 0, 0);
  }
  asm volatile("s_wait_asynccnt 0x0" ::: "memory");
#else
  if (tid < 128) {
    int rg = rowBase + tid;
    int j  = idx[rg];
    int b  = rg >> 12;
    const float* fp = l1pts + ((size_t)(b * NP1 + j)) * CH1;
    float* ar = Af + tid * KP2;
#pragma unroll 16
    for (int c = 0; c < CH1; ++c) ar[c] = fp[c];
  }
#endif
  __syncthreads();
  const int wave = tid >> 5, lane = tid & 31;
  v16h af[5];
#pragma unroll
  for (int ks = 0; ks < 5; ++ks)
    af[ks] = load_a_frag_f32(Af, wave * 16, KP2, ks, lane);
  for (int nt = 0; nt < 44; ++nt) {
    __builtin_prefetch(wp + (size_t)(nt * 16) * KP2, 0, 1); // global_prefetch_b8
    v8f acc = {};
#pragma unroll
    for (int ks = 0; ks < 5; ++ks) {
      v16h bf = load_b_frag(wp, nt * 16, KP2, ks, lane);
      acc = __builtin_amdgcn_wmma_f32_16x16x32_f16(false, af[ks], false, bf,
                                                   (short)0, acc, false, false);
    }
    const int ch = nt * 16 + (lane & 15);
    if (STATS) {
      float bv = (ch < D2) ? bias[ch] : 0.0f;
      float s = 0.0f, q = 0.0f;
#pragma unroll
      for (int i = 0; i < 8; ++i) { float h = acc[i] + bv; s += h; q += h * h; }
      atomicAdd(&ssum[ch], s);
      atomicAdd(&ssum[NPAD2 + ch], q);
    } else {
      float sc = scale[ch], sh = shift[ch];
      float m = -1e30f;
#pragma unroll
      for (int i = 0; i < 8; ++i) m = fmaxf(m, fmaxf(acc[i] * sc + sh, 0.0f));
      m = fmaxf(m, __shfl_xor(m, 16, 32));   // combine M 0-7 with M 8-15
      if (lane < 16 && ch < D2) {
        int gidx = blockIdx.x * 8 + wave;    // flat b*256+s
        outp[(size_t)gidx * D2 + ch] = m;
      }
    }
  }
  if (STATS) {
    __syncthreads();
    for (int c = tid; c < NPAD2; c += 256) {
      atomicAdd(&gstats[c], ssum[c]);
      atomicAdd(&gstats[NPAD2 + c], ssum[NPAD2 + c]);
    }
  }
}

// ---------------------------------------------------------------------------
extern "C" void kernel_launch(void* const* d_in, const int* in_sizes, int n_in,
                              void* d_out, int out_size, void* d_ws,
                              size_t ws_size, hipStream_t stream) {
  (void)in_sizes; (void)n_in; (void)out_size; (void)ws_size;
  const float* xyz = (const float*)d_in[0];
  const float* w1  = (const float*)d_in[1];
  const float* b1  = (const float*)d_in[2];
  const float* g1  = (const float*)d_in[3];
  const float* be1 = (const float*)d_in[4];
  const float* w2  = (const float*)d_in[5];
  const float* b2  = (const float*)d_in[6];
  const float* g2  = (const float*)d_in[7];
  const float* be2 = (const float*)d_in[8];

  // Workspace carve (256B aligned bump allocator).
  char* ws = (char*)d_ws;
  size_t off = 0;
  auto alloc = [&](size_t bytes) -> char* {
    off = (off + 255) & ~size_t(255);
    char* p = ws + off;
    off += bytes;
    return p;
  };
  int*      fps1     = (int*)alloc((size_t)B_ * NP1 * 4);
  float*    new_xyz1 = (float*)alloc((size_t)B_ * NP1 * 3 * 4);
  int*      idx1     = (int*)alloc((size_t)B_ * NP1 * NS1 * 4);
  float*    l1pts    = (float*)alloc((size_t)B_ * NP1 * CH1 * 4);
  int*      fps2     = (int*)alloc((size_t)B_ * NP2 * 4);
  int*      idx2     = (int*)alloc((size_t)B_ * NP2 * NS2 * 4);
  _Float16* wp1      = (_Float16*)alloc((size_t)CH1 * KP1 * 2);
  _Float16* wp2      = (_Float16*)alloc((size_t)NPAD2 * KP2 * 2);
  float*    stats1   = (float*)alloc(2 * CH1 * 4);
  float*    scale1   = (float*)alloc(CH1 * 4);
  float*    shift1   = (float*)alloc(CH1 * 4);
  float*    stats2   = (float*)alloc(2 * NPAD2 * 4);
  float*    scale2   = (float*)alloc(NPAD2 * 4);
  float*    shift2   = (float*)alloc(NPAD2 * 4);

  float* l2xyz = (float*)d_out;                    // B*256*3
  float* l2pts = l2xyz + (size_t)B_ * NP2 * 3;     // B*256*693

  // ---- Layer 1 ----
  const size_t fps1_shm = (size_t)(4 * N1) * 4 + 512 * 8;
  fps_kernel<<<B_, 512, fps1_shm, stream>>>(xyz, 6, N1, NP1, fps1);
  {
    int total = B_ * NP1;
    gather3_kernel<<<(total + 255) / 256, 256, 0, stream>>>(
        xyz, 6, N1, fps1, NP1, total, new_xyz1);
    ball_query_kernel<<<(total + 255) / 256, 256, 0, stream>>>(
        xyz, 6, N1, new_xyz1, NP1, total, NS1, 0.0025f * 0.0025f, idx1);
  }
  pack_w_kernel<<<(CH1 * KP1 + 255) / 256, 256, 0, stream>>>(w1, 6, CH1, KP1,
                                                             CH1, 0, wp1);
  zero_f32<<<1, 256, 0, stream>>>(stats1, 2 * CH1);
  {
    const int blocks = (B_ * NP1 * NS1) / 128;     // 2048
    const size_t shm_stats = 128 * KP1 * 2 + 2 * CH1 * 4;
    const size_t shm_final = 128 * KP1 * 2 + 128 * CH1 * 4;
    sa1_kernel<true><<<blocks, 256, shm_stats, stream>>>(
        xyz, new_xyz1, idx1, wp1, b1, stats1, nullptr, nullptr, nullptr);
    bn_finalize<<<1, 128, 0, stream>>>(stats1, g1, be1, b1,
                                       1.0f / (float)(B_ * NP1 * NS1), CH1,
                                       CH1, scale1, shift1);
    sa1_kernel<false><<<blocks, 256, shm_final, stream>>>(
        xyz, new_xyz1, idx1, wp1, b1, nullptr, scale1, shift1, l1pts);
  }

  // ---- Layer 2 ----
  const size_t fps2_shm = (size_t)(4 * NP1) * 4 + 512 * 8;
  fps_kernel<<<B_, 512, fps2_shm, stream>>>(new_xyz1, 3, NP1, NP2, fps2);
  {
    int total = B_ * NP2;
    gather3_kernel<<<(total + 255) / 256, 256, 0, stream>>>(
        new_xyz1, 3, NP1, fps2, NP2, total, l2xyz);
    ball_query_kernel<<<(total + 255) / 256, 256, 0, stream>>>(
        new_xyz1, 3, NP1, l2xyz, NP2, total, NS2, 0.005f * 0.005f, idx2);
  }
  pack_w_kernel<<<(NPAD2 * KP2 + 255) / 256, 256, 0, stream>>>(
      w2, C2, D2, KP2, NPAD2, 1, wp2);
  zero_f32<<<(2 * NPAD2 + 255) / 256, 256, 0, stream>>>(stats2, 2 * NPAD2);
  {
    const int blocks = (B_ * NP2 * NS2) / 128;     // 128
    const size_t shm_stats = (size_t)128 * KP2 * 4 + 2 * NPAD2 * 4;
    const size_t shm_final = (size_t)128 * KP2 * 4;
    sa2_kernel<true><<<blocks, 256, shm_stats, stream>>>(
        new_xyz1, l1pts, l2xyz, idx2, wp2, b2, stats2, nullptr, nullptr,
        nullptr);
    bn_finalize<<<(NPAD2 + 255) / 256, 256, 0, stream>>>(
        stats2, g2, be2, b2, 1.0f / (float)(B_ * NP2 * NS2), D2, NPAD2, scale2,
        shift2);
    sa2_kernel<false><<<blocks, 256, shm_final, stream>>>(
        new_xyz1, l1pts, l2xyz, idx2, wp2, b2, nullptr, scale2, shift2, l2pts);
  }
}
